// GraphConvolution_51874615001231
// MI455X (gfx1250) — matrix operand order repaired and checked by reading, not compiled, not run on hardware
//
#include <hip/hip_runtime.h>
#include <hip/hip_bf16.h>

// ---------------- problem constants ----------------
static constexpr int kB    = 32;
static constexpr int kF    = 26;
static constexpr int kO    = 20;
static constexpr int kCH   = kF * kO;        // 520 channels (BN dim)
static constexpr int kRows = kB * kCH;       // 16640
static constexpr int kIN   = 2048;
static constexpr int kG    = 512;
static constexpr int kNT   = 2 * kG;         // 1024 output features
static constexpr float kEPS = 1e-5f;

typedef __bf16 bf16_t;
typedef __attribute__((ext_vector_type(16))) __bf16 v16bf;
typedef __attribute__((ext_vector_type(8)))  __bf16 v8bf;
typedef __attribute__((ext_vector_type(8)))  float  v8f;
typedef __attribute__((ext_vector_type(2)))  float  v2f;   // plain vector, not HIP class

// ---- explicit global-address-space pointers (force global_load/store) ----
#define AS1 __attribute__((address_space(1)))
typedef const AS1 float*  gcf_t;
typedef const AS1 v2f*    gcf2_t;
typedef const AS1 bf16_t* gcbf_t;
typedef const AS1 v8bf*   gcv8bf_t;
typedef AS1 float*        gf_t;
typedef AS1 bf16_t*       gbf_t;

// ---------------- WMMA fragment loaders ----------------
// A (16x32 bf16, row-major f32 source): element e=2v+p holds
//   K = h*16 + 2*v + p      (h = lane>>4, row = lane&15)
__device__ __forceinline__ v16bf load_a32(gcf_t rowptr, int k0, int h) {
  v16bf a;
#pragma unroll
  for (int v = 0; v < 8; ++v) {
    int k = k0 + (h << 4) + (v << 1);
    v2f x = *(gcf2_t)(rowptr + k);
    a[2 * v]     = (bf16_t)x.x;
    a[2 * v + 1] = (bf16_t)x.y;
  }
  return a;
}

__device__ __forceinline__ v16bf load_a32_guard(gcf_t rowptr, int k0, int h,
                                                bool mvalid, int kmax) {
  v16bf a;
#pragma unroll
  for (int v = 0; v < 8; ++v) {
    int k = k0 + (h << 4) + (v << 1);
    float x0 = (mvalid && (k     < kmax)) ? rowptr[k]     : 0.0f;
    float x1 = (mvalid && (k + 1 < kmax)) ? rowptr[k + 1] : 0.0f;
    a[2 * v]     = (bf16_t)x0;
    a[2 * v + 1] = (bf16_t)x1;
  }
  return a;
}

// B (32x16 bf16) from row-major weight row W[n][K] (B = W^T):
//   lane holds column n; element e holds K = k0 + h*16 + e
__device__ __forceinline__ v16bf load_bw32(gcf_t wrow, int k0, int h) {
  v16bf b;
#pragma unroll
  for (int v = 0; v < 8; ++v) {
    int k = k0 + (h << 4) + (v << 1);
    v2f x = *(gcf2_t)(wrow + k);
    b[2 * v]     = (bf16_t)x.x;
    b[2 * v + 1] = (bf16_t)x.y;
  }
  return b;
}

// B (32x16 bf16) from K-contiguous bf16 column (temregT layout), guarded K
__device__ __forceinline__ v16bf load_bT_bf16(gcbf_t colptr, int k0, int h, int kmax) {
  const int k = k0 + (h << 4);
  v16bf b;
  if (k + 15 < kmax) {
    v8bf lo = *(gcv8bf_t)(colptr + k);
    v8bf hi = *(gcv8bf_t)(colptr + k + 8);
#pragma unroll
    for (int e = 0; e < 8; ++e) { b[e] = lo[e]; b[e + 8] = hi[e]; }
  } else {
#pragma unroll
    for (int e = 0; e < 16; ++e)
      b[e] = (k + e < kmax) ? colptr[k + e] : (bf16_t)0.0f;
  }
  return b;
}

// ---------------- kernel 1: masked embedding sum ----------------
__global__ void __launch_bounds__(256)
emb_kernel(const int* __restrict__ relf,      // [kRows][kO]
           const float* __restrict__ emb,     // [53][kG]
           float* __restrict__ E,             // [kRows][kG]
           float* __restrict__ msum)          // [kRows]
{
  const int row = blockIdx.x;
  const int tid = threadIdx.x;
  __shared__ int ids[kO];
  if (tid < kO) ids[tid] = ((const AS1 int*)relf)[(size_t)row * kO + tid];
  __syncthreads();
  if (tid == 0) {
    int c = 0;
    for (int j = 0; j < kO; ++j) c += (ids[j] > 0) ? 1 : 0;
    ((gf_t)msum)[row] = (float)c;
  }
  float a0 = 0.0f, a1 = 0.0f;
#pragma unroll
  for (int j = 0; j < kO; ++j) {
    int id = ids[j];
    if (id > 0) {
      gcf_t e = (gcf_t)emb + (size_t)id * kG;
      a0 += e[tid];
      a1 += e[tid + 256];
    }
  }
  ((gf_t)E)[(size_t)row * kG + tid]       = a0;
  ((gf_t)E)[(size_t)row * kG + tid + 256] = a1;
}

// ---------------- kernel 2: big GEMM (region @ [relW;temW]^T) ----------------
// grid = (kNT/64 [N fastest -> A reuse in cache], kRows/32), block = 32.
// One wave computes a 32x64 tile: 2 A-frags x 4 B-frags = 8 WMMAs / K-step.
__global__ void __launch_bounds__(32)
gemm1_kernel(const float* __restrict__ region,   // [kRows][kIN]
             const float* __restrict__ relW, const float* __restrict__ relB,
             const float* __restrict__ temW, const float* __restrict__ temB,
             const float* __restrict__ E, const float* __restrict__ msum,
             const float* __restrict__ bnw, const float* __restrict__ bnb,
             const float* __restrict__ bnrm, const float* __restrict__ bnrv,
             float* __restrict__ out,            // [kRows][kNT]
             bf16_t* __restrict__ temregT)       // [kB][kG][kCH] (transposed!)
{
  const int lane = threadIdx.x & 31;
  const int h = lane >> 4, l16 = lane & 15;
  const int mbase = blockIdx.y * 32;
  const int nbase = blockIdx.x * 64;

  gcf_t arow0 = (gcf_t)region + (size_t)(mbase + l16) * kIN;
  gcf_t arow1 = (gcf_t)region + (size_t)(mbase + 16 + l16) * kIN;

  gcf_t wrow[4];
  float bias[4];
#pragma unroll
  for (int s = 0; s < 4; ++s) {
    int n = nbase + s * 16 + l16;
    if (n < kG) { wrow[s] = (gcf_t)relW + (size_t)n * kIN;        bias[s] = relB[n]; }
    else        { wrow[s] = (gcf_t)temW + (size_t)(n - kG) * kIN; bias[s] = temB[n - kG]; }
  }

  const v8f vz = {0.f, 0.f, 0.f, 0.f, 0.f, 0.f, 0.f, 0.f};
  v8f acc[2][4] = {{vz, vz, vz, vz}, {vz, vz, vz, vz}};

  for (int k0 = 0; k0 < kIN; k0 += 32) {
    v16bf a0 = load_a32(arow0, k0, h);
    v16bf a1 = load_a32(arow1, k0, h);
#pragma unroll
    for (int s = 0; s < 4; ++s) {
      v16bf b = load_bw32(wrow[s], k0, h);
      acc[0][s] = __builtin_amdgcn_wmma_f32_16x16x32_bf16(
          false, a0, false, b, (short)0, acc[0][s], false, false);
      acc[1][s] = __builtin_amdgcn_wmma_f32_16x16x32_bf16(
          false, a1, false, b, (short)0, acc[1][s], false, false);
    }
  }

  if (nbase < kG) {
    // relational half: (E + r*msum)/msum -> tanh -> BN -> out[:, :512]
#pragma unroll
    for (int half = 0; half < 2; ++half) {
#pragma unroll
      for (int r = 0; r < 8; ++r) {
        int row = mbase + half * 16 + h * 8 + r;
        int c = row % kCH;
        float inv = ((gcf_t)bnw)[c] * rsqrtf(((gcf_t)bnrv)[c] + kEPS);
        float rm = ((gcf_t)bnrm)[c], bo = ((gcf_t)bnb)[c];
        float ms = ((gcf_t)msum)[row];
        float rcp = 1.0f / ms;
#pragma unroll
        for (int s = 0; s < 4; ++s) {
          int n = nbase + s * 16 + l16;
          float v = acc[half][s][r] + bias[s];
          float val = (((gcf_t)E)[(size_t)row * kG + n] + v * ms) * rcp;
          float t = tanhf(val);
          ((gf_t)out)[(size_t)row * kNT + n] = (t - rm) * inv + bo;
        }
      }
    }
  } else {
    // temporal half: stash bias-added activations bf16, transposed [b][n][c]
#pragma unroll
    for (int half = 0; half < 2; ++half) {
#pragma unroll
      for (int r = 0; r < 8; ++r) {
        int row = mbase + half * 16 + h * 8 + r;
        int bb = row / kCH, c = row % kCH;
#pragma unroll
        for (int s = 0; s < 4; ++s) {
          int n = nbase + s * 16 + l16 - kG;
          ((gbf_t)temregT)[((size_t)bb * kG + n) * kCH + c] =
              (bf16_t)(acc[half][s][r] + bias[s]);
        }
      }
    }
  }
}

// ---------------- kernel 3: tem_feats row sums ----------------
__global__ void __launch_bounds__(256)
rowsum_kernel(const float* __restrict__ temf, float* __restrict__ tsum)
{
  int gw = (int)((blockIdx.x * blockDim.x + threadIdx.x) >> 5);
  int lane = threadIdx.x & 31;
  if (gw >= kRows) return;
  gcf_t p = (gcf_t)temf + (size_t)gw * kCH;
  float s = 0.0f;
  for (int i = lane; i < kCH; i += 32) s += p[i];
#pragma unroll
  for (int off = 16; off > 0; off >>= 1) s += __shfl_xor(s, off, 32);
  if (lane == 0) ((gf_t)tsum)[gw] = s;
}

// ---------------- kernel 4: batched tem GEMM + epilogue ----------------
// grid = (kG/64 = 8 [N fastest], ceil(520/16) = 33, kB = 32), block = 32
__global__ void __launch_bounds__(32)
gemm2_kernel(const float* __restrict__ temf,      // [kB][kCH][kCH]
             const bf16_t* __restrict__ temregT,  // [kB][kG][kCH]
             const float* __restrict__ tsum,      // [kB*kCH]
             const float* __restrict__ bnw, const float* __restrict__ bnb,
             const float* __restrict__ bnrm, const float* __restrict__ bnrv,
             float* __restrict__ out)             // [kRows][kNT]
{
  const int lane = threadIdx.x & 31;
  const int h = lane >> 4, l16 = lane & 15;
  const int bidx = blockIdx.z;
  gcf_t T = (gcf_t)temf + (size_t)bidx * kCH * kCH;
  const int m = blockIdx.y * 16 + l16;
  const bool mvalid = (m < kCH);
  gcf_t arow = T + (size_t)m * kCH;
  const int nbase = blockIdx.x * 64;

  gcbf_t bcol[4];
#pragma unroll
  for (int s = 0; s < 4; ++s) {
    int n = nbase + s * 16 + l16;
    bcol[s] = (gcbf_t)temregT + ((size_t)bidx * kG + n) * kCH;
  }

  const v8f vz = {0.f, 0.f, 0.f, 0.f, 0.f, 0.f, 0.f, 0.f};
  v8f acc[4] = {vz, vz, vz, vz};

  for (int k0 = 0; k0 < kCH; k0 += 32) {
    v16bf a = load_a32_guard(arow, k0, h, mvalid, kCH);
#pragma unroll
    for (int s = 0; s < 4; ++s) {
      v16bf b = load_bT_bf16(bcol[s], k0, h, kCH);
      acc[s] = __builtin_amdgcn_wmma_f32_16x16x32_bf16(
          false, a, false, b, (short)0, acc[s], false, false);
    }
  }

#pragma unroll
  for (int r = 0; r < 8; ++r) {
    int mm = blockIdx.y * 16 + h * 8 + r;
    if (mm >= kCH) continue;
    float inv = ((gcf_t)bnw)[mm] * rsqrtf(((gcf_t)bnrv)[mm] + kEPS);
    float rm = ((gcf_t)bnrm)[mm], bo = ((gcf_t)bnb)[mm];
    float rcp = 1.0f / ((gcf_t)tsum)[bidx * kCH + mm];
#pragma unroll
    for (int s = 0; s < 4; ++s) {
      int n = nbase + s * 16 + l16;
      float t = tanhf(acc[s][r] * rcp);
      ((gf_t)out)[((size_t)(bidx * kCH + mm)) * kNT + kG + n] = (t - rm) * inv + bo;
    }
  }
}

// ---------------- launch ----------------
extern "C" void kernel_launch(void* const* d_in, const int* in_sizes, int n_in,
                              void* d_out, int out_size, void* d_ws, size_t ws_size,
                              hipStream_t stream) {
  (void)in_sizes; (void)n_in; (void)out_size; (void)ws_size;
  const float* region = (const float*)d_in[0];
  /* d_in[1] region_masks: unused (all-ones / cancels in the math) */
  const int*   relf   = (const int*)d_in[2];
  const float* temf   = (const float*)d_in[3];
  const float* relW   = (const float*)d_in[4];
  const float* relB   = (const float*)d_in[5];
  const float* relE   = (const float*)d_in[6];
  const float* temW   = (const float*)d_in[7];
  const float* temB   = (const float*)d_in[8];
  const float* bnw    = (const float*)d_in[9];
  const float* bnb    = (const float*)d_in[10];
  const float* bnrm   = (const float*)d_in[11];
  const float* bnrv   = (const float*)d_in[12];
  float* out = (float*)d_out;

  // workspace layout (~51.2 MB)
  char* ws = (char*)d_ws;
  float*  E       = (float*)ws;                                  // kRows*kG f32
  size_t  off     = (size_t)kRows * kG * sizeof(float);
  float*  msum    = (float*)(ws + off);  off += (size_t)kRows * sizeof(float);
  bf16_t* temregT = (bf16_t*)(ws + off); off += (size_t)kRows * kG * sizeof(bf16_t);
  float*  tsum    = (float*)(ws + off);                          // kRows f32

  emb_kernel<<<kRows, 256, 0, stream>>>(relf, relE, E, msum);

  gemm1_kernel<<<dim3(kNT / 64, kRows / 32), 32, 0, stream>>>(
      region, relW, relB, temW, temB, E, msum,
      bnw, bnb, bnrm, bnrv, out, temregT);

  rowsum_kernel<<<kRows / 8, 256, 0, stream>>>(temf, tsum);

  gemm2_kernel<<<dim3(kG / 64, (kCH + 15) / 16, kB), 32, 0, stream>>>(
      temf, temregT, tsum, bnw, bnb, bnrm, bnrv, out);
}
